// Net_44392781971913
// MI455X (gfx1250) — matrix operand order repaired and checked by reading, not compiled, not run on hardware
//
#include <hip/hip_runtime.h>
#include <stdint.h>

// ---- CDNA5 vector types for WMMA operands ----
typedef __attribute__((ext_vector_type(2)))  float    v2f;
typedef __attribute__((ext_vector_type(8)))  float    v8f;
typedef __attribute__((ext_vector_type(8)))  _Float16 v8h;
typedef __attribute__((ext_vector_type(16))) _Float16 v16h;

#define BETA   0.95f
#define THRESH 1.0f
#define Bsz 128
#define Tsz 200
#define Isz 784
#define Hsz 1024
#define Osz 10

#define KCHUNK 392        // 784 = 2 chunks of 392 (divisible by 4 -> b128 staging)
#define XS     396        // padded LDS row stride (floats): 396 mod 64 = 12 -> conflict-free b64 A reads
#define SPKS   1032       // padded LDS row stride (halves) for spike matrix
#define NV4    (16 * (KCHUNK / 4))   // 1568 float4 elements per x chunk

__global__ __launch_bounds__(512) void snn_fused_kernel(
    const float* __restrict__ x,  const float* __restrict__ w1,
    const float* __restrict__ b1, const float* __restrict__ w2,
    const float* __restrict__ b2, float* __restrict__ out)
{
    __shared__ float    xlds[16 * XS];       // 25,344 B : x_t tile chunk (16 rows x 392 K)
    __shared__ _Float16 spklds[16 * SPKS];   // 33,024 B : spk1 (16 rows x 1024 H) exact in f16

    const int tid    = threadIdx.x;
    const int lane   = tid & 31;
    const int wave   = tid >> 5;
    const int laneN  = lane & 15;     // N column (B/C/D) or M row (A) within 16x16 tile
    const int laneHi = lane >> 4;     // half-wave selector
    const int bBase  = blockIdx.x * 16;   // 8 blocks x 16 batch rows
    const int hBase  = wave * 64;         // 16 waves x 64 hidden units (4 N-tiles)

    // ---- t-invariant staging descriptors: each thread copies up to 4 float4s ----
    // idx = tid + i*512 over 1568 vec4s; row = idx/98, col4 = (idx%98)*4.
    const float* xbase[4];
    uint32_t     ldsoff[4];
#pragma unroll
    for (int i = 0; i < 4; ++i) {
        int idx = tid + i * 512;
        int g   = (idx < NV4) ? idx : 0;           // clamp; guarded at use
        int row = g / (KCHUNK / 4);
        int c4  = (g % (KCHUNK / 4)) * 4;
        xbase[i]  = x + (size_t)(bBase + row) * Tsz * Isz + c4;
        // wave-relative LDS byte address = low 32 bits of the generic LDS pointer
        ldsoff[i] = (uint32_t)(uintptr_t)&xlds[row * XS + c4];
    }

    // Per-tile w1 element offset for this lane's B fragment (f32 16x16x4 layout:
    // b = { B[2*hi][n], B[2*hi+1][n] } with B[k][n] = w1[h=n][k]) + bias fragment.
    int   wOff[4];
    float b1v[4];
#pragma unroll
    for (int tt = 0; tt < 4; ++tt) {
        int h    = hBase + tt * 16 + laneN;
        wOff[tt] = h * Isz + 2 * laneHi;
        b1v[tt]  = b1[h];
    }

    // Persistent membrane state lives in accumulator VGPRs for all 200 steps.
    v8f mem1[4];
#pragma unroll
    for (int tt = 0; tt < 4; ++tt) {
        v8f z = {0.f,0.f,0.f,0.f,0.f,0.f,0.f,0.f};
        mem1[tt] = z;
    }
    v8f mem2 = {0.f,0.f,0.f,0.f,0.f,0.f,0.f,0.f};

    const float b2v   = (laneN < Osz) ? b2[laneN] : 0.0f;
    const float wmask = (laneN < Osz) ? 1.0f : 0.0f;
    const float* w2row = w2 + (size_t)((laneN < Osz) ? laneN : 0) * Hsz; // clamped, masked below

    float* outSpk = out;
    float* outMem = out + (size_t)Bsz * Tsz * Osz;

    for (int t = 0; t < Tsz; ++t) {
        // Fold leak + subtract-reset + bias into WMMA C-init:
        //   mem1_new = (beta*mem1 - rst*TH + b1) + x_t @ w1^T   accumulated by WMMA
        v8f acc[4];
#pragma unroll
        for (int tt = 0; tt < 4; ++tt) {
#pragma unroll
            for (int j = 0; j < 8; ++j) {
                float m = mem1[tt][j];
                acc[tt][j] = BETA * m - ((m > THRESH) ? THRESH : 0.0f) + b1v[tt];
            }
        }

        for (int c = 0; c < 2; ++c) {
            const int kBase = c * KCHUNK;
            // ---- async stage: x[bBase..+15, t, kBase..+391] -> LDS, no VGPR round-trip ----
#pragma unroll
            for (int i = 0; i < 4; ++i) {
                if (tid + i * 512 < NV4) {
                    const float* p = xbase[i] + (size_t)t * Isz + kBase;
                    asm volatile("global_load_async_to_lds_b128 %0, %1, off"
                                 :: "v"(ldsoff[i]), "v"(p) : "memory");
                    if (t + 1 < Tsz) __builtin_prefetch(p + Isz, 0, 1); // next timestep
                }
            }
            asm volatile("s_wait_asynccnt 0x0" ::: "memory");
            __syncthreads();

            // ---- K loop: 98 steps of f32 WMMA (K=4); A fragment shared across 4 N-tiles ----
            const float* w1c = w1 + kBase;
#pragma unroll 2
            for (int k0 = 0; k0 < KCHUNK; k0 += 4) {
                v2f a = *(const v2f*)&xlds[laneN * XS + k0 + 2 * laneHi];
#pragma unroll
                for (int tt = 0; tt < 4; ++tt) {
                    v2f b = *(const v2f*)(w1c + wOff[tt] + k0);
                    acc[tt] = __builtin_amdgcn_wmma_f32_16x16x4_f32(
                        false, a, false, b, (short)0, acc[tt], false, false);
                }
            }
            __syncthreads();
        }

        // ---- mem1 update + spikes to LDS as f16 (spikes {0,1} are exact) ----
#pragma unroll
        for (int tt = 0; tt < 4; ++tt) {
            mem1[tt] = acc[tt];
#pragma unroll
            for (int j = 0; j < 8; ++j) {
                float s  = (acc[tt][j] > THRESH) ? 1.0f : 0.0f;
                int  row = j + 8 * laneHi;                 // C layout: M = j (+8 for hi half)
                spklds[row * SPKS + hBase + tt * 16 + laneN] = (_Float16)s;
            }
        }
        __syncthreads();

        // ---- layer 2 (O=10 padded to 16): wave 0 only, K=1024 via 32 f16 WMMAs.
        //      Other waves overlap by pre-staging next timestep's x chunk. ----
        if (wave == 0) {
            v8f acc2;
#pragma unroll
            for (int j = 0; j < 8; ++j) {
                float m = mem2[j];
                acc2[j] = BETA * m - ((m > THRESH) ? THRESH : 0.0f) + b2v;
            }
            for (int kb = 0; kb < Hsz; kb += 32) {
                // A = spk1 (16x32 f16): lane m holds K {8*hi..+7} and {16+8*hi..+7}
                v8h lo = *(const v8h*)&spklds[laneN * SPKS + kb + 8 * laneHi];
                v8h hi = *(const v8h*)&spklds[laneN * SPKS + kb + 16 + 8 * laneHi];
                v16h A = __builtin_shufflevector(lo, hi,
                         0,1,2,3,4,5,6,7,8,9,10,11,12,13,14,15);
                // B = w2^T (32x16 f16): lane n holds K {8*hi..+7} and {16+8*hi..+7} of column n
                v16h Bm;
                int  ks = kb + 8 * laneHi;
#pragma unroll
                for (int e = 0; e < 16; ++e)
                    Bm[e] = (_Float16)(w2row[ks + e + (e & 8)] * wmask);
                acc2 = __builtin_amdgcn_wmma_f32_16x16x32_f16(
                    false, A, false, Bm, (short)0, acc2, false, false);
            }
            mem2 = acc2;
            if (laneN < Osz) {
#pragma unroll
                for (int j = 0; j < 8; ++j) {
                    int    row = j + 8 * laneHi;  // batch row within tile
                    size_t o   = ((size_t)(bBase + row) * Tsz + t) * Osz + laneN;
                    outSpk[o] = (acc2[j] > THRESH) ? 1.0f : 0.0f;
                    outMem[o] = acc2[j];
                }
            }
        }
        // No barrier needed here: next-t staging writes xlds (distinct from spklds);
        // spklds is rewritten only after the next timestep's layer-1 barriers,
        // which wave 0 also participates in.
    }
}

extern "C" void kernel_launch(void* const* d_in, const int* in_sizes, int n_in,
                              void* d_out, int out_size, void* d_ws, size_t ws_size,
                              hipStream_t stream) {
    const float* x  = (const float*)d_in[0];
    const float* w1 = (const float*)d_in[1];
    const float* b1 = (const float*)d_in[2];
    const float* w2 = (const float*)d_in[3];
    const float* b2 = (const float*)d_in[4];
    snn_fused_kernel<<<dim3(Bsz / 16), dim3(512), 0, stream>>>(
        x, w1, b1, w2, b2, (float*)d_out);
}
